// MusicXSkyline_4002909520703
// MI455X (gfx1250) — compile-verified, matrix operand order checked by reading, not run on hardware
//
#include <hip/hip_runtime.h>
#include <hip/hip_bf16.h>

#define Bsz 64
#define Ssz 65536
#define Fsz 6
#define TILE 2048
#define NT (Ssz / TILE)        // 32 tiles per batch row
#define THREADS 256            // 8 wave32s
#define PER_TH (TILE / THREADS) // 8 tokens per thread
#define NWAVES (THREADS / 32)

typedef __attribute__((ext_vector_type(16))) _Float16 v16h;
typedef __attribute__((ext_vector_type(8)))  float    v8f;
typedef unsigned int v4u __attribute__((ext_vector_type(4)));
typedef int          v8i __attribute__((ext_vector_type(8)));
typedef int          v4i __attribute__((ext_vector_type(4)));

// ---------------------------------------------------------------------------
// Kernel 1: per-tile keep-count. The count reduction runs through
// v_wmma_f32_16x16x32_f16: A = this wave's 512 A-slots holding the 256 keep
// flags (0/1, exact in f16) + zero padding, B = all-ones. Since B is uniform,
// sum(D) = 16 * sum(flags) regardless of the WMMA register layout -> exact.
// ---------------------------------------------------------------------------
__global__ void k_count(const int* __restrict__ gid, const int* __restrict__ flag,
                        int* __restrict__ tileCounts) {
    __shared__ int waveCnt[NWAVES];
    const int bidx = blockIdx.x;
    const int b    = bidx / NT;
    const int tile = bidx % NT;
    const int* g  = gid  + (size_t)b * Ssz;
    const int* fl = flag + (size_t)b * Ssz;
    const int base = tile * TILE + threadIdx.x * PER_TH;

    int gv[PER_TH + 1];
#pragma unroll
    for (int i = 0; i < PER_TH; ++i) gv[i] = g[base + i];
    gv[PER_TH] = (base + PER_TH < Ssz) ? g[base + PER_TH] : -1; // sentinel: last token is group-last

    v16h a;
#pragma unroll
    for (int i = 0; i < 16; ++i) a[i] = (_Float16)0.0f;
#pragma unroll
    for (int i = 0; i < PER_TH; ++i) {
        const int fv   = fl[gv[i]];
        const int keep = ((fv == 1) | (gv[i] != gv[i + 1])) ? 1 : 0;
        a[i] = (_Float16)(float)keep;
    }

    v16h ones;
#pragma unroll
    for (int i = 0; i < 16; ++i) ones[i] = (_Float16)1.0f;
    v8f c = {};
    v8f d = __builtin_amdgcn_wmma_f32_16x16x32_f16(
        /*neg_a=*/false, a, /*neg_b=*/false, ones,
        /*c_mod=*/(short)0, c, /*reuse_a=*/false, /*reuse_b=*/false);

    float partial = d[0] + d[1] + d[2] + d[3] + d[4] + d[5] + d[6] + d[7];
#pragma unroll
    for (int off = 16; off >= 1; off >>= 1)
        partial += __shfl_xor(partial, off, 32);
    // partial == 16 * (wave's keep count); exact in f32.
    const int wsum = (int)(partial * 0.0625f + 0.5f);

    const int lane = threadIdx.x & 31;
    const int wid  = threadIdx.x >> 5;
    if (lane == 0) waveCnt[wid] = wsum;
    __syncthreads();
    if (threadIdx.x == 0) {
        int r = 0;
#pragma unroll
        for (int w = 0; w < NWAVES; ++w) r += waveCnt[w];
        tileCounts[bidx] = r;
    }
}

// ---------------------------------------------------------------------------
// Kernel 2: per-batch exclusive scan of tile counts (tiny).
// ---------------------------------------------------------------------------
__global__ void k_scan(const int* __restrict__ tileCounts,
                       int* __restrict__ tileOffsets, int* __restrict__ outLen) {
    const int b = threadIdx.x;
    if (b < Bsz) {
        int r = 0;
        for (int t = 0; t < NT; ++t) {
            tileOffsets[b * NT + t] = r;
            r += tileCounts[b * NT + t];
        }
        outLen[b] = r;
    }
}

// ---------------------------------------------------------------------------
// Kernel 3: stage src tile global->LDS via the Tensor Data Mover, block scan
// of keep flags (wave shuffles + LDS), scatter kept rows LDS->global.
// ---------------------------------------------------------------------------
__global__ void k_scatter(const float* __restrict__ src, const int* __restrict__ gid,
                          const int* __restrict__ flag, const int* __restrict__ tileOffsets,
                          float* __restrict__ out) {
    __shared__ float lds_src[TILE * Fsz]; // 48 KB
    __shared__ int waveTot[NWAVES];
    __shared__ int waveOff[NWAVES];

    const int bidx = blockIdx.x;
    const int b    = bidx / NT;
    const int tile = bidx % NT;
    const int* g  = gid  + (size_t)b * Ssz;
    const int* fl = flag + (size_t)b * Ssz;
    const float* gsrc_tile = src + ((size_t)b * Ssz + (size_t)tile * TILE) * Fsz;

#if __has_builtin(__builtin_amdgcn_tensor_load_to_lds) && __has_builtin(__builtin_amdgcn_s_wait_tensorcnt)
    if (threadIdx.x < 32) { // wave 0 issues the TDM op (EXEC ignored by TDM)
        typedef __attribute__((address_space(3))) float* lds_fp;
        const unsigned lds_base = (unsigned)(unsigned long long)(lds_fp)lds_src;
        const unsigned long long ga = (unsigned long long)(const void*)gsrc_tile;
        const unsigned ND = (unsigned)(TILE * Fsz); // 12288 dwords = 48 KB

        // D# group 0: count=1 valid descriptor, lds_addr, 57-bit global addr, type=2.
        v4u g0 = { 1u, lds_base, (unsigned)ga, (unsigned)(ga >> 32) | 0x80000000u };
        // D# group 1: data_size=2 (4B), tensor_dim0=ND, tensor_dim1=1,
        // tile_dim0=ND, tile_dim1=1 (1 row), stride0=ND, no padding/multicast.
        v8i g1;
        g1[0] = 0x00020000;                                  // data_size = 4 bytes
        g1[1] = (int)((ND & 0xFFFFu) << 16);                 // tensor_dim0[15:0]
        g1[2] = (int)((1u << 16) | (ND >> 16));              // tensor_dim1[15:0] | tensor_dim0[31:16]
        g1[3] = (int)((ND & 0xFFFFu) << 16);                 // tile_dim0 | tensor_dim1[31:16]
        g1[4] = 1;                                           // tile_dim1=1, tile_dim2=0
        g1[5] = (int)ND;                                     // tensor_dim0_stride[31:0]
        g1[6] = 0;
        g1[7] = 0;
        v4i g2 = { 0, 0, 0, 0 };
        v4i g3 = { 0, 0, 0, 0 };
        v8i g4 = { 0, 0, 0, 0, 0, 0, 0, 0 };                 // extra group (6-arg form), unused
        __builtin_amdgcn_tensor_load_to_lds(g0, g1, g2, g3, g4, 0);
        __builtin_amdgcn_s_wait_tensorcnt(0);
    }
    __syncthreads();
#else
    for (int i = threadIdx.x; i < TILE * Fsz; i += blockDim.x) lds_src[i] = gsrc_tile[i];
    __syncthreads();
#endif

    const int base = tile * TILE + threadIdx.x * PER_TH;
    int gv[PER_TH + 1];
#pragma unroll
    for (int i = 0; i < PER_TH; ++i) gv[i] = g[base + i];
    gv[PER_TH] = (base + PER_TH < Ssz) ? g[base + PER_TH] : -1;

    int keep[PER_TH];
    int tsum = 0;
#pragma unroll
    for (int i = 0; i < PER_TH; ++i) {
        const int fv = fl[gv[i]];
        keep[i] = ((fv == 1) | (gv[i] != gv[i + 1])) ? 1 : 0;
        tsum += keep[i];
    }

    // wave32 inclusive scan of per-thread sums
    const int lane = threadIdx.x & 31;
    const int wid  = threadIdx.x >> 5;
    int v = tsum;
#pragma unroll
    for (int dlt = 1; dlt < 32; dlt <<= 1) {
        const int n = __shfl_up(v, dlt, 32);
        if (lane >= dlt) v += n;
    }
    if (lane == 31) waveTot[wid] = v;
    __syncthreads();
    if (threadIdx.x == 0) {
        int r = 0;
#pragma unroll
        for (int w = 0; w < NWAVES; ++w) { waveOff[w] = r; r += waveTot[w]; }
    }
    __syncthreads();

    int pos = tileOffsets[bidx] + waveOff[wid] + (v - tsum); // exclusive prefix for this thread
#pragma unroll
    for (int i = 0; i < PER_TH; ++i) {
        if (keep[i]) {
            float* o = out + ((size_t)b * Ssz + (size_t)pos) * Fsz;
            const float* srow = &lds_src[(threadIdx.x * PER_TH + i) * Fsz];
#pragma unroll
            for (int f = 0; f < Fsz; ++f) o[f] = srow[f];
            ++pos;
        }
    }
}

// ---------------------------------------------------------------------------
// Kernel 4: zero the padded tail rows, write the length mask, zero n_beats.
// ---------------------------------------------------------------------------
__global__ void k_finish(const int* __restrict__ outLen, float* __restrict__ out,
                         float* __restrict__ mask, float* __restrict__ nbeats) {
    const int b = blockIdx.x;
    const int len = outLen[b];
    for (int s = threadIdx.x; s < Ssz; s += blockDim.x) {
        mask[(size_t)b * Ssz + s] = (s < len) ? 1.0f : 0.0f;
        if (s >= len) {
            float* o = out + ((size_t)b * Ssz + s) * Fsz;
#pragma unroll
            for (int f = 0; f < Fsz; ++f) o[f] = 0.0f;
        }
    }
    if (b == 0 && threadIdx.x < Bsz) nbeats[threadIdx.x] = 0.0f;
}

extern "C" void kernel_launch(void* const* d_in, const int* in_sizes, int n_in,
                              void* d_out, int out_size, void* d_ws, size_t ws_size,
                              hipStream_t stream) {
    const float* src  = (const float*)d_in[0]; // [B,S,F] f32
    const int*   gid  = (const int*)d_in[1];   // [B,S] i32 (sorted per row)
    const int*   flag = (const int*)d_in[2];   // [B,S] i32

    float* out    = (float*)d_out;                     // [B,S,F]
    float* mask   = out  + (size_t)Bsz * Ssz * Fsz;    // [B,S] as 0/1 floats
    float* nbeats = mask + (size_t)Bsz * Ssz;          // [B]

    int* tileCounts  = (int*)d_ws;
    int* tileOffsets = tileCounts  + Bsz * NT;
    int* outLen      = tileOffsets + Bsz * NT;

    k_count  <<<Bsz * NT, THREADS, 0, stream>>>(gid, flag, tileCounts);
    k_scan   <<<1, Bsz, 0, stream>>>(tileCounts, tileOffsets, outLen);
    k_scatter<<<Bsz * NT, THREADS, 0, stream>>>(src, gid, flag, tileOffsets, out);
    k_finish <<<Bsz, 1024, 0, stream>>>(outLen, out, mask, nbeats);
}
